// QuantLinear_24721831756583
// MI455X (gfx1250) — compile-verified, matrix operand order checked by reading, not run on hardware
//
#include <hip/hip_runtime.h>

// ---------------------------------------------------------------------------
// Q4_K quantized linear:  out[M=4096, N=4096] = X[M,K=4096] * W[N,K]^T
// Phase 1: dequantize W (4-bit superblocks) -> f16, convert X fp32 -> f16.
//          Both 32 MB operands then live in MI455X's 192 MB L2.
// Phase 2: double-buffered LDS WMMA GEMM. Global->LDS staging uses CDNA5
//          async LDS-fill instructions (global_load_async_to_lds_b128,
//          ASYNCcnt) so tile loads bypass VGPRs and overlap the WMMA work.
// ---------------------------------------------------------------------------

typedef __attribute__((ext_vector_type(16))) _Float16 v16h;
typedef __attribute__((ext_vector_type(8)))  float    v8f;

#define OUT_F  4096
#define IN_F   4096
#define M_TOK  4096
#define NUMEL  (OUT_F * IN_F)

union FragH {
    v16h   h;
    float4 f4[2];
};

// ---- CDNA5 async global->LDS copy (ASYNCcnt-tracked, no VGPR round trip) ---
__device__ __forceinline__ void async_fill_b128(unsigned lds_addr, const void* gaddr) {
    // GV form: vdst = LDS byte offset, v[addr] = 64-bit global address
    asm volatile("global_load_async_to_lds_b128 %0, %1, off"
                 :: "v"(lds_addr), "v"(gaddr) : "memory");
}
__device__ __forceinline__ void wait_async0() {
    asm volatile("s_wait_asynccnt 0x0" ::: "memory");
}

// ---------------------------------------------------------------------------
// Dequantize Q4_K -> f16.  One thread handles 4 packed bytes = 8 weights,
// always inside one 32-weight subblock.
//   w = q * (d*scale/(15*63)) + (d*min/63 + dmin)
// ---------------------------------------------------------------------------
__global__ __launch_bounds__(256)
void dequant_q4k(const int* __restrict__ packed,
                 const float* __restrict__ d,
                 const float* __restrict__ dmin,
                 const int* __restrict__ scales,
                 const int* __restrict__ mins,
                 _Float16* __restrict__ w_h) {
    const int t = blockIdx.x * blockDim.x + threadIdx.x;
    const long wBase = (long)t * 8;
    if (wBase >= (long)NUMEL) return;

    const int sup = (int)(wBase >> 8);
    const int sub = (int)((wBase >> 5) & 7);

    const float ds = d[sup];
    const float sc = ds * (float)scales[sup * 8 + sub] * (1.0f / (15.0f * 63.0f));
    const float of = ds * (float)mins[sup * 8 + sub] * (1.0f / 63.0f) + dmin[sup];

    const int4 p = *(const int4*)(packed + t * 4);
    const int b[4] = {p.x, p.y, p.z, p.w};

    union { _Float16 h[8]; float4 f4; } o;
#pragma unroll
    for (int j = 0; j < 4; ++j) {
        const int lo = b[j] & 15;
        const int hi = (b[j] >> 4) & 15;
        o.h[2 * j + 0] = (_Float16)((float)lo * sc + of);
        o.h[2 * j + 1] = (_Float16)((float)hi * sc + of);
    }
    *(float4*)(w_h + wBase) = o.f4;
}

// ---------------------------------------------------------------------------
// Convert activations fp32 -> f16, 8 elements per thread.
// ---------------------------------------------------------------------------
__global__ __launch_bounds__(256)
void cvt_x_f16(const float* __restrict__ x, _Float16* __restrict__ x_h) {
    const long t = (long)blockIdx.x * blockDim.x + threadIdx.x;
    const long base = t * 8;
    const float4 a = *(const float4*)(x + base);
    const float4 b = *(const float4*)(x + base + 4);
    union { _Float16 h[8]; float4 f4; } o;
    o.h[0] = (_Float16)a.x; o.h[1] = (_Float16)a.y;
    o.h[2] = (_Float16)a.z; o.h[3] = (_Float16)a.w;
    o.h[4] = (_Float16)b.x; o.h[5] = (_Float16)b.y;
    o.h[6] = (_Float16)b.z; o.h[7] = (_Float16)b.w;
    *(float4*)(x_h + base) = o.f4;
}

// ---------------------------------------------------------------------------
// WMMA GEMM: block = 256 threads = 8 wave32s, block tile 128x128, K-step 64,
// double-buffered LDS filled by async LDS-load instructions.
// Wave (wm 0..3, wn 0..1) owns a 32x64 C slab = 2x4 WMMA tiles.
// ---------------------------------------------------------------------------
#define BM  128
#define BN  128
#define BK  64
#define LDK (BK + 8)   // padded row stride in halves (spreads LDS banks)

__global__ __launch_bounds__(256)
void gemm_wmma_f16(const _Float16* __restrict__ A,   // [M_TOK, IN_F]
                   const _Float16* __restrict__ B,   // [OUT_F, IN_F]
                   float* __restrict__ C) {          // [M_TOK, OUT_F]
    __shared__ _Float16 As[2][BM * LDK];
    __shared__ _Float16 Bs[2][BN * LDK];

    const int tid  = threadIdx.x;
    const int lane = tid & 31;
    const int wave = tid >> 5;     // 0..7
    const int wm   = wave & 3;     // 32-row slab within block tile
    const int wn   = wave >> 2;    // 64-col slab within block tile

    const int m0 = blockIdx.y * BM;
    const int n0 = blockIdx.x * BN;

    v8f acc[2][4];
#pragma unroll
    for (int mt = 0; mt < 2; ++mt)
#pragma unroll
        for (int nt = 0; nt < 4; ++nt) {
            v8f z = {};
            acc[mt][nt] = z;
        }

    // Copy mapping: 4 threads per row, 16 halves (32 B = 2 x b128) each,
    // two row-passes cover 128 rows -> 8 async b128 fills per thread/stage.
    const int crow = tid >> 2;         // 0..63
    const int ccol = (tid & 3) * 16;   // 0,16,32,48

    const int r = lane & 15;           // row within 16x16 tile
    const int h = lane >> 4;           // K-half selector

    auto issue = [&](int sb, int k0) {
#pragma unroll
        for (int p = 0; p < 2; ++p) {
            const _Float16* ga = A + (long)(m0 + crow + p * 64) * IN_F + k0 + ccol;
            const _Float16* gb = B + (long)(n0 + crow + p * 64) * IN_F + k0 + ccol;
            const unsigned la = (unsigned)(uintptr_t)&As[sb][(crow + p * 64) * LDK + ccol];
            const unsigned lb = (unsigned)(uintptr_t)&Bs[sb][(crow + p * 64) * LDK + ccol];
#pragma unroll
            for (int c = 0; c < 2; ++c) {
                async_fill_b128(la + c * 16, ga + c * 8);   // 16 B per lane
                async_fill_b128(lb + c * 16, gb + c * 8);
            }
        }
    };

    auto compute = [&](int sb) {
#pragma unroll
        for (int kk = 0; kk < 2; ++kk) {
            const int kb = kk * 32 + h * 8;

            FragH aF[2];
#pragma unroll
            for (int mt = 0; mt < 2; ++mt) {
                const _Float16* pA = &As[sb][(wm * 32 + mt * 16 + r) * LDK + kb];
                aF[mt].f4[0] = *(const float4*)(pA);
                aF[mt].f4[1] = *(const float4*)(pA + 16);
            }
            FragH bF[4];
#pragma unroll
            for (int nt = 0; nt < 4; ++nt) {
                const _Float16* pB = &Bs[sb][(wn * 64 + nt * 16 + r) * LDK + kb];
                bF[nt].f4[0] = *(const float4*)(pB);
                bF[nt].f4[1] = *(const float4*)(pB + 16);
            }

#pragma unroll
            for (int mt = 0; mt < 2; ++mt)
#pragma unroll
                for (int nt = 0; nt < 4; ++nt)
                    acc[mt][nt] = __builtin_amdgcn_wmma_f32_16x16x32_f16(
                        /*neg_a=*/false, aF[mt].h,
                        /*neg_b=*/false, bF[nt].h,
                        /*c_mod=*/(short)0, acc[mt][nt],
                        /*reuse_a=*/false, /*reuse_b=*/false);
        }
    };

    // ---- prologue: fill buffer 0 ----
    issue(0, 0);

    // ---- main loop: 2 stages per iteration so buffer index is literal ----
#pragma unroll 1
    for (int k0 = 0; k0 < IN_F; k0 += 2 * BK) {
        wait_async0();          // own stage-(i) fills complete
        __syncthreads();        // all waves' fills complete; prior reads done
        if (k0 + BK < IN_F) issue(1, k0 + BK);   // overlap with compute(0)
        compute(0);

        wait_async0();
        __syncthreads();
        if (k0 + 2 * BK < IN_F) issue(0, k0 + 2 * BK);
        compute(1);
    }

    // ---- epilogue: C/D layout -> lane r = col, VGPR j + 8*h = row ----
#pragma unroll
    for (int mt = 0; mt < 2; ++mt) {
#pragma unroll
        for (int nt = 0; nt < 4; ++nt) {
            const int row0 = m0 + wm * 32 + mt * 16 + h * 8;
            const int col  = n0 + wn * 64 + nt * 16 + r;
#pragma unroll
            for (int j = 0; j < 8; ++j)
                C[(long)(row0 + j) * OUT_F + col] = acc[mt][nt][j];
        }
    }
}

// ---------------------------------------------------------------------------
// Host-side launch
// ---------------------------------------------------------------------------
extern "C" void kernel_launch(void* const* d_in, const int* in_sizes, int n_in,
                              void* d_out, int out_size, void* d_ws, size_t ws_size,
                              hipStream_t stream) {
    const float* x      = (const float*)d_in[0];   // [2,2048,4096] fp32
    const int*   packed = (const int*)d_in[1];     // [NUMEL/2] bytes-in-int32
    const float* d      = (const float*)d_in[2];   // [65536]
    const float* dmin   = (const float*)d_in[3];   // [65536]
    const int*   scales = (const int*)d_in[4];     // [524288]
    const int*   mins   = (const int*)d_in[5];     // [524288]
    float*       out    = (float*)d_out;           // [2,2048,4096] fp32

    _Float16* w_h = (_Float16*)d_ws;               // 32 MB
    _Float16* x_h = w_h + (size_t)NUMEL;           // 32 MB

    dequant_q4k<<<NUMEL / 8 / 256, 256, 0, stream>>>(packed, d, dmin, scales, mins, w_h);
    cvt_x_f16<<<(M_TOK * IN_F) / 8 / 256, 256, 0, stream>>>(x, x_h);

    dim3 grid(OUT_F / BN, M_TOK / BM);   // 32 x 32 blocks
    gemm_wmma_f16<<<grid, 256, 0, stream>>>(x_h, w_h, out);
}
